// SelfAttention_22634477650084
// MI455X (gfx1250) — matrix operand order repaired and checked by reading, not compiled, not run on hardware
//
#include <hip/hip_runtime.h>
#include <hip/hip_bf16.h>

#define EMBED 512
#define HEADS 8
#define HDIM  64
#define SEQ   4096
#define NB    2

typedef __attribute__((ext_vector_type(16))) _Float16 v16h;
typedef __attribute__((ext_vector_type(8)))  _Float16 v8h;
typedef __attribute__((ext_vector_type(8)))  float    v8f;
typedef __attribute__((ext_vector_type(4)))  float    v4f;

union AF { v16h v; _Float16 h[16]; v8h q2[2]; };

// ---- WMMA wrapper: D = A(16x32 f16) * B(32x16 f16) + C(16x16 f32) ----
__device__ __forceinline__ v8f wmma16(v16h a, v16h b, v8f c) {
    return __builtin_amdgcn_wmma_f32_16x16x32_f16(
        /*neg_a=*/false, a, /*neg_b=*/false, b,
        /*c_mod=*/(short)0, c, /*reuse_a=*/false, /*reuse_b=*/false);
}

// A fragment (16x32, M x K) from row-major f16 tile; per ISA table:
// lane: m=lane%16, c=lane/16; halfs at [m][kb+c*8 .. +7] and [m][kb+16+c*8 .. +7]
__device__ __forceinline__ v16h load_a_f16(const _Float16* base, int ld, int kb) {
    const int lane = threadIdx.x & 31;
    const int m = lane & 15, c = lane >> 4;
    const _Float16* p = base + (size_t)m * ld + kb + c * 8;
    AF f;
    f.q2[0] = *reinterpret_cast<const v8h*>(p);
    f.q2[1] = *reinterpret_cast<const v8h*>(p + 16);
    return f.v;
}

// B fragment (32x16, K x N) from row-major B^T tile (rows = N, cols = K):
// lane: n=lane%16, c=lane/16; halfs at [n][kb+c*8 .. +7] and [n][kb+16+c*8 .. +7]
__device__ __forceinline__ v16h load_b_f16(const _Float16* base, int ld, int kb) {
    const int lane = threadIdx.x & 31;
    const int n = lane & 15, c = lane >> 4;
    const _Float16* p = base + (size_t)n * ld + kb + c * 8;
    AF f;
    f.q2[0] = *reinterpret_cast<const v8h*>(p);
    f.q2[1] = *reinterpret_cast<const v8h*>(p + 16);
    return f.v;
}

// A fragment loaded from f32 source with on-the-fly f16 conversion
__device__ __forceinline__ v16h load_a_f32(const float* base, int ld, int kb) {
    const int lane = threadIdx.x & 31;
    const int m = lane & 15, c = lane >> 4;
    const float* p = base + (size_t)m * ld + kb + c * 8;
    v4f x0 = *reinterpret_cast<const v4f*>(p);
    v4f x1 = *reinterpret_cast<const v4f*>(p + 4);
    v4f x2 = *reinterpret_cast<const v4f*>(p + 16);
    v4f x3 = *reinterpret_cast<const v4f*>(p + 20);
    AF f;
#pragma unroll
    for (int i = 0; i < 4; ++i) {
        f.h[i]      = (_Float16)x0[i];
        f.h[4 + i]  = (_Float16)x1[i];
        f.h[8 + i]  = (_Float16)x2[i];
        f.h[12 + i] = (_Float16)x3[i];
    }
    return f.v;
}

// ---- CDNA5 async copy: 16B/lane global -> LDS, tracked by ASYNCcnt ----
__device__ __forceinline__ void async_copy_b128(uint32_t lds_byte_off,
                                                const _Float16* gsrc) {
    asm volatile("global_load_async_to_lds_b128 %0, %1, off"
                 :
                 : "v"(lds_byte_off), "v"((uint64_t)(uintptr_t)gsrc)
                 : "memory");
}
__device__ __forceinline__ void wait_async0() {
    asm volatile("s_wait_asynccnt 0" ::: "memory");
}

// ---------------- stage 0: f32 -> f16 weight conversion ----------------
__global__ void cvt_f16_kernel(const float* __restrict__ src,
                               _Float16* __restrict__ dst, int n) {
    int i = blockIdx.x * blockDim.x + threadIdx.x;
    if (i < n) dst[i] = (_Float16)src[i];
}

// ---------------- stage 1: per-head projections ----------------
// blockIdx.y: 0 -> Q, 1 -> K, 2 -> V(transposed output)
__global__ __launch_bounds__(256) void proj_kernel(
    const float* __restrict__ xq, const float* __restrict__ xk,
    const float* __restrict__ xv,
    const _Float16* __restrict__ Wq16, const _Float16* __restrict__ Wk16,
    const _Float16* __restrict__ Wv16,
    _Float16* __restrict__ Qp, _Float16* __restrict__ Kp,
    _Float16* __restrict__ Vt) {
    const int wave = threadIdx.x >> 5;
    const int lane = threadIdx.x & 31;
    const int nn = lane & 15, c = lane >> 4;

    const float* x; const _Float16* W; _Float16* dst; int transpose;
    if (blockIdx.y == 0)      { x = xq; W = Wq16; dst = Qp; transpose = 0; }
    else if (blockIdx.y == 1) { x = xk; W = Wk16; dst = Kp; transpose = 0; }
    else                      { x = xv; W = Wv16; dst = Vt; transpose = 1; }

    const int tile = blockIdx.x * 8 + wave;          // 0 .. N*H*(SEQ/16)-1
    const int qt = tile & (SEQ / 16 - 1);
    const int nh = tile / (SEQ / 16);
    const int h = nh & (HEADS - 1);
    const int n = nh >> 3;

    const float* xbase = x + ((size_t)(n * SEQ + qt * 16)) * EMBED + h * HDIM;
    v8f acc[4] = {v8f{}, v8f{}, v8f{}, v8f{}};
#pragma unroll
    for (int kb = 0; kb < HDIM; kb += 32) {
        v16h a = load_a_f32(xbase, EMBED, kb);
#pragma unroll
        for (int t = 0; t < 4; ++t) {
            v16h b = load_b_f16(W + (size_t)(16 * t) * HDIM, HDIM, kb);
            acc[t] = wmma16(a, b, acc[t]);
        }
    }
    _Float16* obase = dst + ((size_t)(n * HEADS + h)) * SEQ * HDIM;
#pragma unroll
    for (int t = 0; t < 4; ++t)
#pragma unroll
        for (int r = 0; r < 8; ++r) {
            const int tok = qt * 16 + r + 8 * c;
            const int d = 16 * t + nn;
            const _Float16 val = (_Float16)acc[t][r];
            if (!transpose) obase[(size_t)tok * HDIM + d] = val;
            else            obase[(size_t)d * SEQ + tok]  = val;
        }
}

// ---------------- stage 2: flash attention (async double-buffered) --------
// block = 4 waves, one (n,h,64-row q block); each wave owns 16 q rows.
// K/V^T 64x64 tiles stream into LDS via ASYNCcnt-tracked b128 copies.
__global__ __launch_bounds__(128) void attn_kernel(
    const _Float16* __restrict__ Qp, const _Float16* __restrict__ Kp,
    const _Float16* __restrict__ Vt, const int* __restrict__ mask,
    _Float16* __restrict__ attn_out) {
    __shared__ _Float16 ldsK[2][64 * 64];     // [token][d], 8KB each
    __shared__ _Float16 ldsV[2][64 * 64];     // [d][token], 8KB each
    __shared__ _Float16 ldsP[4][16 * 64];     // per-wave P tile

    const int wave = threadIdx.x >> 5;
    const int lane = threadIdx.x & 31;
    const int nn = lane & 15, c = lane >> 4;
    const int tid = threadIdx.x;

    const int qb = blockIdx.x & (SEQ / 64 - 1);
    const int nh = blockIdx.x / (SEQ / 64);
    const int h = nh & (HEADS - 1);
    const int n = nh >> 3;

    const size_t head_off = (size_t)(n * HEADS + h) * SEQ * HDIM;
    const _Float16* Qh = Qp + head_off;
    const _Float16* Kh = Kp + head_off;
    const _Float16* Vh = Vt + head_off;
    const int q0 = qb * 64 + wave * 16;

    // issue async copies for one 64-token K tile + V^T tile into buffer `buf`
    auto copy_tiles = [&](int buf, int k0) {
        const _Float16* gk = Kh + (size_t)k0 * HDIM;   // contiguous 8KB
        const uint32_t kofs = (uint32_t)(uintptr_t)&ldsK[buf][0];
#pragma unroll
        for (int j = 0; j < 4; ++j) {
            const int cidx = tid + 128 * j;            // 0..511 x 16B
            async_copy_b128(kofs + (uint32_t)cidx * 16, gk + cidx * 8);
        }
        const uint32_t vofs = (uint32_t)(uintptr_t)&ldsV[buf][0];
#pragma unroll
        for (int j = 0; j < 4; ++j) {
            const int cidx = tid + 128 * j;
            const int row = cidx >> 3;                 // d 0..63
            const int col = cidx & 7;                  // 16B chunk in row
            async_copy_b128(vofs + (uint32_t)(row * 64 + col * 8) * 2,
                            Vh + (size_t)row * SEQ + k0 + col * 8);
        }
    };

    const v16h qa0 = load_a_f16(Qh + (size_t)q0 * HDIM, HDIM, 0);
    const v16h qa1 = load_a_f16(Qh + (size_t)q0 * HDIM, HDIM, 32);

    v8f o[4] = {v8f{}, v8f{}, v8f{}, v8f{}};
    float mrow[8], lrow[8];
#pragma unroll
    for (int r = 0; r < 8; ++r) { mrow[r] = -3.0e38f; lrow[r] = 0.0f; }

    const float scale = 0.04419417382415922f;        // 1/sqrt(EMBED)
    const float LOG2E = 1.4426950408889634f;
    const int* mptr = mask + n * SEQ;
    _Float16* Pt = ldsP[wave];

    // prologue: stream in tile 0
    copy_tiles(0, 0);
    wait_async0();
    __syncthreads();

    for (int kt = 0; kt < SEQ / 64; ++kt) {
        const int cur = kt & 1;
        const int k0 = kt * 64;
        // prefetch next tile while computing this one
        if (kt + 1 < SEQ / 64) copy_tiles(cur ^ 1, (kt + 1) * 64);

        const _Float16* Ktile = ldsK[cur];
        const _Float16* Vtile = ldsV[cur];

        // ---- S = Q K^T (16x64 tile, 8 WMMAs; B-frags from LDS) ----
        v8f s[4] = {v8f{}, v8f{}, v8f{}, v8f{}};
#pragma unroll
        for (int t = 0; t < 4; ++t) {
            const _Float16* kb_base = Ktile + (size_t)(16 * t) * HDIM;
            s[t] = wmma16(qa0, load_b_f16(kb_base, HDIM, 0), s[t]);
            s[t] = wmma16(qa1, load_b_f16(kb_base, HDIM, 32), s[t]);
        }
        // ---- scale + mask ----
#pragma unroll
        for (int t = 0; t < 4; ++t) {
            const int mk = mptr[k0 + 16 * t + nn];
#pragma unroll
            for (int r = 0; r < 8; ++r) {
                const float v = s[t][r] * scale;
                s[t][r] = (mk == 0) ? -1.0e20f : v;
            }
        }
        // ---- online softmax (row reduce across 16-lane half-waves) ----
#pragma unroll
        for (int r = 0; r < 8; ++r) {
            float mx = fmaxf(fmaxf(s[0][r], s[1][r]), fmaxf(s[2][r], s[3][r]));
            mx = fmaxf(mx, __shfl_xor(mx, 1, 32));
            mx = fmaxf(mx, __shfl_xor(mx, 2, 32));
            mx = fmaxf(mx, __shfl_xor(mx, 4, 32));
            mx = fmaxf(mx, __shfl_xor(mx, 8, 32));
            const float mnew = fmaxf(mrow[r], mx);
            const float alpha = exp2f((mrow[r] - mnew) * LOG2E);
            mrow[r] = mnew;
            float rs = 0.0f;
#pragma unroll
            for (int t = 0; t < 4; ++t) {
                const float p = exp2f((s[t][r] - mnew) * LOG2E);
                s[t][r] = p;
                rs += p;
            }
            rs += __shfl_xor(rs, 1, 32);
            rs += __shfl_xor(rs, 2, 32);
            rs += __shfl_xor(rs, 4, 32);
            rs += __shfl_xor(rs, 8, 32);
            lrow[r] = lrow[r] * alpha + rs;
#pragma unroll
            for (int t = 0; t < 4; ++t) o[t][r] *= alpha;
        }
        // ---- P -> LDS (C-layout scatter), then reload as A-fragments ----
#pragma unroll
        for (int t = 0; t < 4; ++t)
#pragma unroll
            for (int r = 0; r < 8; ++r)
                Pt[(r + 8 * c) * 64 + 16 * t + nn] = (_Float16)s[t][r];
        asm volatile("s_wait_dscnt 0" ::: "memory");
        // ---- O += P V (8 WMMAs; B-frags from LDS V^T tile) ----
        const v16h pa0 = load_a_f16(Pt, 64, 0);
        const v16h pa1 = load_a_f16(Pt, 64, 32);
#pragma unroll
        for (int t = 0; t < 4; ++t) {
            const _Float16* vb = Vtile + (size_t)(16 * t) * 64;
            o[t] = wmma16(pa0, load_b_f16(vb, 64, 0), o[t]);
            o[t] = wmma16(pa1, load_b_f16(vb, 64, 32), o[t]);
        }
        // double-buffer handoff: next tile landed, everyone done with cur
        wait_async0();
        __syncthreads();
    }
    // ---- epilogue: O /= l, store f16 [n][q][h*64+d] ----
    _Float16* obase = attn_out + ((size_t)(n * SEQ) + q0) * EMBED + h * HDIM;
#pragma unroll
    for (int r = 0; r < 8; ++r) {
        const float inv = 1.0f / lrow[r];
#pragma unroll
        for (int t = 0; t < 4; ++t)
            obase[(size_t)(r + 8 * c) * EMBED + 16 * t + nn] =
                (_Float16)(o[t][r] * inv);
    }
}

// ---------------- stage 3: out = attn @ Wo^T + bo ----------------
__global__ __launch_bounds__(256) void outproj_kernel(
    const _Float16* __restrict__ attn, const _Float16* __restrict__ Wo16,
    const float* __restrict__ bo, float* __restrict__ out) {
    const int wave = threadIdx.x >> 5;
    const int lane = threadIdx.x & 31;
    const int nn = lane & 15, c = lane >> 4;

    const int tile = blockIdx.x * 8 + wave;   // 512 row-tiles x 8 col-tiles
    const int ct = tile & 7;
    const int rt = tile >> 3;

    const _Float16* abase = attn + (size_t)rt * 16 * EMBED;
    v8f acc[4] = {v8f{}, v8f{}, v8f{}, v8f{}};
#pragma unroll 4
    for (int kc = 0; kc < EMBED / 32; ++kc) {
        v16h a = load_a_f16(abase, EMBED, kc * 32);
#pragma unroll
        for (int t = 0; t < 4; ++t) {
            v16h b = load_b_f16(Wo16 + (size_t)(ct * 64 + 16 * t) * EMBED,
                                EMBED, kc * 32);
            acc[t] = wmma16(a, b, acc[t]);
        }
    }
    float* obase = out + (size_t)rt * 16 * EMBED + ct * 64;
#pragma unroll
    for (int t = 0; t < 4; ++t) {
        const float bias = bo[ct * 64 + 16 * t + nn];
#pragma unroll
        for (int r = 0; r < 8; ++r)
            obase[(size_t)(r + 8 * c) * EMBED + 16 * t + nn] = acc[t][r] + bias;
    }
}

// ---------------- host ----------------
extern "C" void kernel_launch(void* const* d_in, const int* in_sizes, int n_in,
                              void* d_out, int out_size, void* d_ws, size_t ws_size,
                              hipStream_t stream) {
    (void)in_sizes; (void)n_in; (void)out_size; (void)ws_size;
    const float* values = (const float*)d_in[0];
    const float* keys_  = (const float*)d_in[1];
    const float* query  = (const float*)d_in[2];
    const int*   mask   = (const int*)d_in[3];
    const float* Wv     = (const float*)d_in[4];
    const float* Wk     = (const float*)d_in[5];
    const float* Wq     = (const float*)d_in[6];
    const float* Wo     = (const float*)d_in[7];
    const float* bo     = (const float*)d_in[8];
    float* out = (float*)d_out;

    _Float16* ws = (_Float16*)d_ws;
    const size_t NTOK = (size_t)NB * HEADS * SEQ * HDIM;   // 4,194,304 halfs
    _Float16* Wq16 = ws + 0;
    _Float16* Wk16 = ws + 4096;
    _Float16* Wv16 = ws + 8192;
    _Float16* Wo16 = ws + 12288;             // 512*512 halfs
    _Float16* Qp   = ws + 274432;
    _Float16* Kp   = Qp + NTOK;
    _Float16* Vt   = Kp + NTOK;
    _Float16* Ao   = Vt + NTOK;

    cvt_f16_kernel<<<16, 256, 0, stream>>>(Wq, Wq16, HDIM * HDIM);
    cvt_f16_kernel<<<16, 256, 0, stream>>>(Wk, Wk16, HDIM * HDIM);
    cvt_f16_kernel<<<16, 256, 0, stream>>>(Wv, Wv16, HDIM * HDIM);
    cvt_f16_kernel<<<1024, 256, 0, stream>>>(Wo, Wo16, EMBED * EMBED);

    proj_kernel<<<dim3(512, 3), 256, 0, stream>>>(query, keys_, values,
                                                  Wq16, Wk16, Wv16, Qp, Kp, Vt);
    attn_kernel<<<NB * HEADS * (SEQ / 64), 128, 0, stream>>>(Qp, Kp, Vt, mask, Ao);
    outproj_kernel<<<512, 256, 0, stream>>>(Ao, Wo16, bo, out);
}
